// Attention_44590350467732
// MI455X (gfx1250) — matrix operand order repaired and checked by reading, not compiled, not run on hardware
//
#include <hip/hip_runtime.h>
#include <hip/hip_bf16.h>
#include <hip/hip_fp16.h>

// ---------------------------------------------------------------------------
// Problem constants (from reference)
// ---------------------------------------------------------------------------
#define DIM     1024
#define HEADS   32
#define DHEAD   32
#define NTOK    65      // tokens per window
#define NPAD    80      // token dim padded to 5 x 16
#define KPADP   96      // K padding for attn@v (3 x 32)
#define NWINB   512     // total windows (B)
#define BH      (NWINB * HEADS)   // 16384
#define ROWS    (NWINB * NTOK)    // 33280  (multiple of 32)
#define MTILES  (ROWS / 16)       // 2080
#define MPAIRS  (ROWS / 32)       // 1040 (32-row wave tiles)
#define KC      (DIM / 32)        // 32 K-chunks of 32
#define NQKV    3072
#define NTQKV   (NQKV / 16)       // 192 n-tiles
#define NTOUT   (DIM / 16)        // 64 n-tiles

typedef __attribute__((ext_vector_type(16))) _Float16 v16h;
typedef __attribute__((ext_vector_type(8)))  _Float16 v8h;
typedef __attribute__((ext_vector_type(8)))  float    v8f;

static __device__ inline v8f wmma16(v16h a, v16h b, v8f c) {
  // D = A(16x32 f16) x B(32x16 f16) + C(16x16 f32)
  return __builtin_amdgcn_wmma_f32_16x16x32_f16(false, a, false, b, (short)0, c,
                                                false, false);
}

// Load a 16x32 f16 A-fragment from a row-major matrix.
// rowptr = start of row (m_tile*16 + lane%16); contiguous K order:
// lanes 0-15 hold K {0..15}, lanes 16-31 hold K {16..31}.
static __device__ inline v16h load_a_frag(const _Float16* rowptr, int kbase,
                                          bool hiHalf) {
  const _Float16* p = rowptr + kbase + (hiHalf ? 16 : 0);
  v16h a;
  v8h* ap = reinterpret_cast<v8h*>(&a);
  ap[0] = *reinterpret_cast<const v8h*>(p);
  ap[1] = *reinterpret_cast<const v8h*>(p + 8);
  return a;
}

// ---------------------------------------------------------------------------
// FiLM MLP (tiny): h1 = silu(cond @ w1 + b1) ; gb = h1 @ w2 + b2
// ---------------------------------------------------------------------------
__global__ __launch_bounds__(256) void film1_kernel(
    const float* __restrict__ cond, const float* __restrict__ w1,
    const float* __restrict__ b1, float* __restrict__ h1) {
  int r = blockIdx.x;                       // 0..15
  const float* cr = cond + (size_t)r * 512;
  for (int c = threadIdx.x; c < 2048; c += blockDim.x) {
    float acc = b1[c];
    for (int k = 0; k < 512; ++k) acc += cr[k] * w1[(size_t)k * 2048 + c];
    h1[(size_t)r * 2048 + c] = acc / (1.0f + __expf(-acc));   // silu
  }
}

__global__ __launch_bounds__(256) void film2_kernel(
    const float* __restrict__ h1, const float* __restrict__ w2,
    const float* __restrict__ b2, float* __restrict__ gb) {
  int r = blockIdx.x;                       // 0..15
  const float* hr = h1 + (size_t)r * 2048;
  for (int c = threadIdx.x; c < 2048; c += blockDim.x) {
    float acc = b2[c];
    for (int k = 0; k < 2048; ++k) acc += hr[k] * w2[(size_t)k * 2048 + c];
    gb[(size_t)r * 2048 + c] = acc;
  }
}

// ---------------------------------------------------------------------------
// Pack w_qkv (1024x3072) and w_out (1024x1024) into f16 WMMA B-fragments.
// Layout: [n_tile][k_chunk][lane][16 halves]; lane holds column N = nt*16+lane%16,
// 16 contiguous K values starting at kc*32 (+16 for lanes 16-31).
// ---------------------------------------------------------------------------
__global__ __launch_bounds__(256) void pack_w_kernel(
    const float* __restrict__ wqkv, const float* __restrict__ wout,
    _Float16* __restrict__ pq, _Float16* __restrict__ po) {
  int gid  = blockIdx.x * blockDim.x + threadIdx.x;
  int lane = gid & 31;
  int f    = gid >> 5;
  const int FQ = NTQKV * KC;   // 6144
  const int FO = NTOUT * KC;   // 2048
  if (f < FQ) {
    int nt = f / KC, kc = f % KC;
    int N  = nt * 16 + (lane & 15);
    int K0 = kc * 32 + ((lane & 16) ? 16 : 0);
    _Float16* dst = pq + ((size_t)f * 32 + lane) * 16;
    for (int i = 0; i < 16; ++i)
      dst[i] = (_Float16)wqkv[(size_t)(K0 + i) * NQKV + N];
  } else if (f < FQ + FO) {
    int f2 = f - FQ;
    int nt = f2 / KC, kc = f2 % KC;
    int N  = nt * 16 + (lane & 15);
    int K0 = kc * 32 + ((lane & 16) ? 16 : 0);
    _Float16* dst = po + ((size_t)f2 * 32 + lane) * 16;
    for (int i = 0; i < 16; ++i)
      dst[i] = (_Float16)wout[(size_t)(K0 + i) * DIM + N];
  }
}

// ---------------------------------------------------------------------------
// LayerNorm (no affine) + FiLM (gamma/beta per image) -> f16 activations
// one block per row of 1024
// ---------------------------------------------------------------------------
__global__ __launch_bounds__(256) void ln_film_kernel(
    const float* __restrict__ x, const float* __restrict__ gb,
    _Float16* __restrict__ xnh) {
  int row = blockIdx.x;
  int tid = threadIdx.x;
  const float* xr = x + (size_t)row * DIM;
  float v[4], s = 0.f, s2 = 0.f;
#pragma unroll
  for (int i = 0; i < 4; ++i) {
    float t = xr[tid + 256 * i];
    v[i] = t; s += t; s2 += t * t;
  }
#pragma unroll
  for (int m = 1; m <= 16; m <<= 1) {
    s  += __shfl_xor(s,  m, 32);
    s2 += __shfl_xor(s2, m, 32);
  }
  __shared__ float rs[8], rs2[8];
  if ((tid & 31) == 0) { rs[tid >> 5] = s; rs2[tid >> 5] = s2; }
  __syncthreads();
  s = 0.f; s2 = 0.f;
#pragma unroll
  for (int w = 0; w < 8; ++w) { s += rs[w]; s2 += rs2[w]; }
  float mean = s * (1.0f / DIM);
  float var  = s2 * (1.0f / DIM) - mean * mean;
  float inv  = rsqrtf(var + 1e-5f);
  int bimg = row / (NTOK * 32);   // window b = row/65, image = b/32 => row/2080
  const float* g  = gb + (size_t)bimg * 2048;
  const float* be = g + 1024;
#pragma unroll
  for (int i = 0; i < 4; ++i) {
    int c = tid + 256 * i;
    float y = (v[i] - mean) * inv * g[c] + be[c];
    xnh[(size_t)row * DIM + c] = (_Float16)y;
  }
}

// ---------------------------------------------------------------------------
// QKV GEMM: (33280 x 1024) x (1024 x 3072), wave tile = 32 x 128
// (2 A fragments share 8 B fragments -> 1.25 loads per WMMA).
// Epilogue scatters into head-major layouts:
//   q,k -> [bh][n(pad80)][32]   v -> [bh][32][n(pad80)] (transposed)
// ---------------------------------------------------------------------------
__global__ __launch_bounds__(256) void qkv_gemm_kernel(
    const _Float16* __restrict__ xnh, const _Float16* __restrict__ wpk,
    _Float16* __restrict__ qh, _Float16* __restrict__ kh,
    _Float16* __restrict__ vt) {
  const int NS = NTQKV / 8;                 // 24 strips of 128 columns
  int wid  = blockIdx.x * 8 + (threadIdx.x >> 5);
  int lane = threadIdx.x & 31;
  int lo   = lane & 15;
  bool hiHalf = (lane & 16) != 0;
  int mp = wid / NS;                        // 32-row tile index
  int ns = wid % NS;
  if (mp >= MPAIRS) return;

  v8f acc[2][8];
#pragma unroll
  for (int m = 0; m < 2; ++m)
#pragma unroll
    for (int t = 0; t < 8; ++t)
#pragma unroll
      for (int i = 0; i < 8; ++i) acc[m][t][i] = 0.f;

  const _Float16* arow0 = xnh + (size_t)(mp * 32 + lo) * DIM;
  const _Float16* arow1 = arow0 + (size_t)16 * DIM;
  const _Float16* bbase = wpk + ((size_t)(ns * 8) * KC * 32 + lane) * 16;

  for (int kc = 0; kc < KC; ++kc) {
    v16h a0 = load_a_frag(arow0, kc * 32, hiHalf);
    v16h a1 = load_a_frag(arow1, kc * 32, hiHalf);
    const _Float16* bp = bbase + (size_t)kc * 32 * 16;
    // warm next K-chunk of B (global_prefetch_b8)
    __builtin_prefetch(bp + 32 * 16, 0, 3);
#pragma unroll
    for (int t = 0; t < 8; ++t) {
      v16h b = *reinterpret_cast<const v16h*>(bp + (size_t)t * KC * 32 * 16);
      acc[0][t] = wmma16(a0, b, acc[0][t]);
      acc[1][t] = wmma16(a1, b, acc[1][t]);
    }
  }

#pragma unroll
  for (int t = 0; t < 8; ++t) {
    int N   = ns * 128 + t * 16 + lo;
    int sel = N >> 10;                      // 0=q 1=k 2=v
    int c   = N & 1023;
    int h   = c >> 5, d = c & 31;
#pragma unroll
    for (int m = 0; m < 2; ++m)
#pragma unroll
      for (int r = 0; r < 8; ++r) {
        int M = mp * 32 + m * 16 + r + (hiHalf ? 8 : 0);
        int b = M / NTOK;
        int n = M - b * NTOK;
        int bh = b * HEADS + h;
        _Float16 hv = (_Float16)acc[m][t][r];
        if (sel == 0)      qh[((size_t)bh * NPAD + n) * DHEAD + d] = hv;
        else if (sel == 1) kh[((size_t)bh * NPAD + n) * DHEAD + d] = hv;
        else               vt[((size_t)bh * DHEAD + d) * NPAD + n] = hv;
      }
  }
}

// ---------------------------------------------------------------------------
// RMS-norm q and k per (bh, token) over dhead=32, with gamma inputs.
// ---------------------------------------------------------------------------
__global__ __launch_bounds__(256) void rmsnorm_qk_kernel(
    _Float16* __restrict__ qh, _Float16* __restrict__ kh,
    const float* __restrict__ qg, const float* __restrict__ kg) {
  int idx = blockIdx.x * blockDim.x + threadIdx.x;
  if (idx >= BH * NTOK) return;
  int bh = idx / NTOK, n = idx % NTOK;
  int h  = bh & (HEADS - 1);
  _Float16* q = qh + ((size_t)bh * NPAD + n) * DHEAD;
  _Float16* k = kh + ((size_t)bh * NPAD + n) * DHEAD;
  float tq[DHEAD], tk[DHEAD], sq = 0.f, sk = 0.f;
#pragma unroll
  for (int d = 0; d < DHEAD; ++d) {
    tq[d] = (float)q[d]; sq += tq[d] * tq[d];
    tk[d] = (float)k[d]; sk += tk[d] * tk[d];
  }
  const float fs = 5.656854249492380f;      // sqrt(32)
  float iq = fs / fmaxf(sqrtf(sq), 1e-12f);
  float ik = fs / fmaxf(sqrtf(sk), 1e-12f);
#pragma unroll
  for (int d = 0; d < DHEAD; ++d) {
    q[d] = (_Float16)(tq[d] * iq * qg[h * DHEAD + d]);
    k[d] = (_Float16)(tk[d] * ik * kg[h * DHEAD + d]);
  }
}

// ---------------------------------------------------------------------------
// Attention: one block per (window, head); 5 waves = 5 query M-tiles.
// sim = q k^T (+bias), softmax over keys, out = p v. Output f16 row-major.
// ---------------------------------------------------------------------------
__global__ __launch_bounds__(160) void attn_kernel(
    const _Float16* __restrict__ qh, const _Float16* __restrict__ kh,
    const _Float16* __restrict__ vt, const float* __restrict__ rel_emb,
    const int* __restrict__ rel_idx, _Float16* __restrict__ outh) {
  int bh   = blockIdx.x;
  int b    = bh >> 5;
  int h    = bh & (HEADS - 1);
  int mt   = threadIdx.x >> 5;              // 0..4
  int lane = threadIdx.x & 31;
  int lo   = lane & 15;
  bool hiHalf = (lane & 16) != 0;

  __shared__ __align__(32) _Float16 pbuf[5][16][KPADP];

  const _Float16* qb = qh + (size_t)bh * NPAD * DHEAD;
  const _Float16* kb = kh + (size_t)bh * NPAD * DHEAD;
  const _Float16* vb = vt + (size_t)bh * DHEAD * NPAD;

  // Q fragment: rows mt*16..+15, K = dhead = 32
  v16h aq = load_a_frag(qb + (size_t)(mt * 16 + lo) * DHEAD, 0, hiHalf);

  // sim tiles: B fragment = k rows (column-major access is contiguous per lane)
  v8f acc[5];
#pragma unroll
  for (int nt = 0; nt < 5; ++nt) {
    v8f z;
#pragma unroll
    for (int i = 0; i < 8; ++i) z[i] = 0.f;
    v16h bk = *reinterpret_cast<const v16h*>(
        kb + (size_t)(nt * 16 + lo) * DHEAD + (hiHalf ? 16 : 0));
    acc[nt] = wmma16(aq, bk, z);
  }

  // relative position bias + key padding mask
#pragma unroll
  for (int nt = 0; nt < 5; ++nt) {
    int j = nt * 16 + lo;
#pragma unroll
    for (int r = 0; r < 8; ++r) {
      int i = mt * 16 + r + (hiHalf ? 8 : 0);
      if (j < NTOK) {
        if (i < NTOK)
          acc[nt][r] += rel_emb[(size_t)rel_idx[i * NTOK + j] * HEADS + h];
      } else {
        acc[nt][r] = -1e30f;
      }
    }
  }

  // softmax: each D VGPR r holds one row across 16 lanes (per half-wave)
#pragma unroll
  for (int r = 0; r < 8; ++r) {
    float m = acc[0][r];
#pragma unroll
    for (int nt = 1; nt < 5; ++nt) m = fmaxf(m, acc[nt][r]);
#pragma unroll
    for (int s = 1; s <= 8; s <<= 1) m = fmaxf(m, __shfl_xor(m, s, 32));
    float sum = 0.f;
#pragma unroll
    for (int nt = 0; nt < 5; ++nt) {
      float e = __expf(acc[nt][r] - m);
      acc[nt][r] = e; sum += e;
    }
#pragma unroll
    for (int s = 1; s <= 8; s <<= 1) sum += __shfl_xor(sum, s, 32);
    float inv = 1.0f / sum;
#pragma unroll
    for (int nt = 0; nt < 5; ++nt) acc[nt][r] *= inv;
  }

  // transpose probabilities through LDS into A-fragment layout
#pragma unroll
  for (int nt = 0; nt < 5; ++nt) {
    int j = nt * 16 + lo;
#pragma unroll
    for (int r = 0; r < 8; ++r)
      pbuf[mt][r + (hiHalf ? 8 : 0)][j] = (_Float16)acc[nt][r];
  }
  {   // zero K pad columns 80..95
    int j = NPAD + lo;
#pragma unroll
    for (int r = 0; r < 8; ++r)
      pbuf[mt][r + (hiHalf ? 8 : 0)][j] = (_Float16)0.f;
  }
  __syncthreads();

  // out = p @ v : M=16, K=96 (3 chunks), N=32 (2 tiles)
  v8f ao[2];
#pragma unroll
  for (int t = 0; t < 2; ++t)
#pragma unroll
    for (int i = 0; i < 8; ++i) ao[t][i] = 0.f;

#pragma unroll
  for (int kc = 0; kc < 3; ++kc) {
    v16h pa = load_a_frag(&pbuf[mt][lo][0], kc * 32, hiHalf);
#pragma unroll
    for (int t = 0; t < 2; ++t) {
      v16h bv;
      if (kc == 2 && hiHalf) {
#pragma unroll
        for (int i = 0; i < 16; ++i) bv[i] = (_Float16)0.f;   // K 80..95 pad
      } else {
        bv = *reinterpret_cast<const v16h*>(
            vb + (size_t)(t * 16 + lo) * NPAD + kc * 32 + (hiHalf ? 16 : 0));
      }
      ao[t] = wmma16(pa, bv, ao[t]);
    }
  }

#pragma unroll
  for (int t = 0; t < 2; ++t)
#pragma unroll
    for (int r = 0; r < 8; ++r) {
      int i = mt * 16 + r + (hiHalf ? 8 : 0);
      if (i < NTOK)
        outh[((size_t)b * NTOK + i) * DIM + h * DHEAD + t * 16 + lo] =
            (_Float16)ao[t][r];
    }
}

// ---------------------------------------------------------------------------
// Output projection: (33280 x 1024) x (1024 x 1024) -> f32 d_out
// wave tile = 32 x 128
// ---------------------------------------------------------------------------
__global__ __launch_bounds__(256) void out_gemm_kernel(
    const _Float16* __restrict__ ah, const _Float16* __restrict__ wpk,
    float* __restrict__ out) {
  const int NS = NTOUT / 8;                 // 8 strips of 128 columns
  int wid  = blockIdx.x * 8 + (threadIdx.x >> 5);
  int lane = threadIdx.x & 31;
  int lo   = lane & 15;
  bool hiHalf = (lane & 16) != 0;
  int mp = wid / NS;
  int ns = wid % NS;
  if (mp >= MPAIRS) return;

  v8f acc[2][8];
#pragma unroll
  for (int m = 0; m < 2; ++m)
#pragma unroll
    for (int t = 0; t < 8; ++t)
#pragma unroll
      for (int i = 0; i < 8; ++i) acc[m][t][i] = 0.f;

  const _Float16* arow0 = ah + (size_t)(mp * 32 + lo) * DIM;
  const _Float16* arow1 = arow0 + (size_t)16 * DIM;
  const _Float16* bbase = wpk + ((size_t)(ns * 8) * KC * 32 + lane) * 16;

  for (int kc = 0; kc < KC; ++kc) {
    v16h a0 = load_a_frag(arow0, kc * 32, hiHalf);
    v16h a1 = load_a_frag(arow1, kc * 32, hiHalf);
    const _Float16* bp = bbase + (size_t)kc * 32 * 16;
    __builtin_prefetch(bp + 32 * 16, 0, 3);
#pragma unroll
    for (int t = 0; t < 8; ++t) {
      v16h b = *reinterpret_cast<const v16h*>(bp + (size_t)t * KC * 32 * 16);
      acc[0][t] = wmma16(a0, b, acc[0][t]);
      acc[1][t] = wmma16(a1, b, acc[1][t]);
    }
  }

#pragma unroll
  for (int t = 0; t < 8; ++t) {
    int N = ns * 128 + t * 16 + lo;
#pragma unroll
    for (int m = 0; m < 2; ++m)
#pragma unroll
      for (int r = 0; r < 8; ++r) {
        int M = mp * 32 + m * 16 + r + (hiHalf ? 8 : 0);
        out[(size_t)M * DIM + N] = acc[m][t][r];
      }
  }
}

// ---------------------------------------------------------------------------
// Host launcher
// ---------------------------------------------------------------------------
extern "C" void kernel_launch(void* const* d_in, const int* in_sizes, int n_in,
                              void* d_out, int out_size, void* d_ws,
                              size_t ws_size, hipStream_t stream) {
  (void)in_sizes; (void)n_in; (void)out_size; (void)ws_size;
  const float* x      = (const float*)d_in[0];
  const float* cond   = (const float*)d_in[1];
  const float* fw1    = (const float*)d_in[2];
  const float* fb1    = (const float*)d_in[3];
  const float* fw2    = (const float*)d_in[4];
  const float* fb2    = (const float*)d_in[5];
  const float* wqkv   = (const float*)d_in[6];
  const float* qg     = (const float*)d_in[7];
  const float* kg     = (const float*)d_in[8];
  const float* relemb = (const float*)d_in[9];
  const float* wout   = (const float*)d_in[10];
  const int*   relidx = (const int*)d_in[11];
  float* out = (float*)d_out;

  char* ws = (char*)d_ws;
  size_t off = 0;
  auto take = [&](size_t bytes) -> char* {
    char* p = ws + off;
    off += (bytes + 255) & ~(size_t)255;
    return p;
  };
  float*    h1  = (float*)take((size_t)16 * 2048 * 4);
  float*    gb  = (float*)take((size_t)16 * 2048 * 4);
  _Float16* pq  = (_Float16*)take((size_t)DIM * NQKV * 2);
  _Float16* po  = (_Float16*)take((size_t)DIM * DIM * 2);
  _Float16* qh  = (_Float16*)take((size_t)BH * NPAD * DHEAD * 2);
  _Float16* kh  = (_Float16*)take((size_t)BH * NPAD * DHEAD * 2);
  _Float16* vt  = (_Float16*)take((size_t)BH * DHEAD * NPAD * 2);
  _Float16* xnh = (_Float16*)take((size_t)ROWS * DIM * 2);
  _Float16* aoh = xnh;   // attention output reuses the activation buffer
                         // (fully overwritten after the QKV GEMM consumed it)

  film1_kernel<<<16, 256, 0, stream>>>(cond, fw1, fb1, h1);
  film2_kernel<<<16, 256, 0, stream>>>(h1, fw2, fb2, gb);
  pack_w_kernel<<<1024, 256, 0, stream>>>(wqkv, wout, pq, po);
  ln_film_kernel<<<ROWS, 256, 0, stream>>>(x, gb, xnh);
  qkv_gemm_kernel<<<(MPAIRS * (NTQKV / 8) + 7) / 8, 256, 0, stream>>>(
      xnh, pq, qh, kh, vt);
  rmsnorm_qk_kernel<<<(BH * NTOK + 255) / 256, 256, 0, stream>>>(qh, kh, qg, kg);
  attn_kernel<<<BH, 160, 0, stream>>>(qh, kh, vt, relemb, relidx, aoh);
  out_gemm_kernel<<<MPAIRS, 256, 0, stream>>>(aoh, po, out);
}
